// fastAPPNPTransformerBlock_52037823758878
// MI455X (gfx1250) — matrix operand order repaired and checked by reading, not compiled, not run on hardware
//
#include <hip/hip_runtime.h>

typedef __attribute__((ext_vector_type(16))) __bf16 v16bf;
typedef __attribute__((ext_vector_type(8)))  __bf16 v8bf;
typedef __attribute__((ext_vector_type(8)))  float  v8f;

#define FDIM 256
#define HIDN 128
#define NHEAD 8
#define HD 16
#define NCLS 64
#define NB 44
#define KHOPS 10
#define RATIO 0.15075567f   /* 44^-0.5 */

__device__ __forceinline__ __bf16 f2bf(float f) {
  union { float f; unsigned u; } v; v.f = f;
  unsigned r = v.u + 0x7FFFu + ((v.u >> 16) & 1u);
  unsigned short h = (unsigned short)(r >> 16);
  return __builtin_bit_cast(__bf16, h);
}
__device__ __forceinline__ float eluf(float x) {
  return x > 0.f ? x : (__expf(x) - 1.f);
}
__device__ __forceinline__ void atomicMaxFloat(float* addr, float val) {
  unsigned* ua = (unsigned*)addr;
  unsigned old = *ua;
  while (true) {
    float f = __uint_as_float(old);
    if (f >= val) break;
    unsigned assumed = old;
    old = atomicCAS(ua, assumed, __float_as_uint(val));
    if (old == assumed) break;
  }
}

// ---- CDNA5 async global->LDS copy (GLOBAL_LOAD_ASYNC_TO_LDS_B128, ASYNCcnt)
__device__ __forceinline__ unsigned lds_off32(const void* p) {
  // generic shared-aperture address: low 32 bits are the LDS byte offset
  return (unsigned)(size_t)p;
}
__device__ __forceinline__ void async_ld_b128(unsigned lds, const void* g) {
  asm volatile("global_load_async_to_lds_b128 %0, %1, off"
               :: "v"(lds), "v"(g) : "memory");
}
__device__ __forceinline__ void wait_async0() {
#if __has_builtin(__builtin_amdgcn_s_wait_asynccnt)
  __builtin_amdgcn_s_wait_asynccnt(0);
#else
  asm volatile("s_wait_asynccnt 0" ::: "memory");
#endif
}

// ---------------------------------------------------------------- converts
__global__ __launch_bounds__(256) void k_cvt_data(const float* __restrict__ x,
                                                  __bf16* __restrict__ y, size_t n) {
  size_t i = (size_t)blockIdx.x * 256 + threadIdx.x;
  if (i < n) y[i] = f2bf(x[i]);
}

// pack q|k|v|lin1 into WT[512][256] (transposed, bf16) + bias[512]
__global__ __launch_bounds__(256) void k_pack_w(
    const float* __restrict__ qw, const float* __restrict__ kw,
    const float* __restrict__ vw, const float* __restrict__ l1w,
    const float* __restrict__ qb, const float* __restrict__ kb,
    const float* __restrict__ vb, const float* __restrict__ l1b,
    __bf16* __restrict__ WT, float* __restrict__ bias) {
  int j = blockIdx.x;          // 0..511 output column
  int src = j >> 7, jj = j & 127;
  const float* w = (src == 0) ? qw : (src == 1) ? kw : (src == 2) ? vw : l1w;
  const float* b = (src == 0) ? qb : (src == 1) ? kb : (src == 2) ? vb : l1b;
  for (int i = threadIdx.x; i < FDIM; i += 256)
    WT[(size_t)j * FDIM + i] = f2bf(w[(size_t)i * HIDN + jj]);
  if (threadIdx.x == 0) bias[j] = b[jj];
}

__global__ __launch_bounds__(256) void k_pack_w2(const float* __restrict__ w2,
                                                 const float* __restrict__ b2,
                                                 __bf16* __restrict__ WT2,
                                                 float* __restrict__ bias2) {
  int c = blockIdx.x;  // 0..63
  for (int i = threadIdx.x; i < FDIM; i += 256)
    WT2[(size_t)c * FDIM + i] = f2bf(w2[(size_t)i * NCLS + c]);
  if (threadIdx.x == 0) bias2[c] = b2[c];
}

__global__ __launch_bounds__(256) void k_zero_small(float* hm, float* kc, float* ctx) {
  int t = blockIdx.x * 256 + threadIdx.x;
  if (t < NHEAD) hm[t] = -3.4e38f;
  if (t < NHEAD * NB) kc[t] = 0.f;
  if (t < NHEAD * NB * HD) ctx[t] = 0.f;
}

__global__ __launch_bounds__(256) void k_zero_f(float* p, int n) {
  int i = blockIdx.x * 256 + threadIdx.x;
  if (i < n) p[i] = 0.f;
}

// ------------------------------------------------- WMMA GEMM  out = X @ WT^T + b
// X: [n_rows,256] bf16 row-major; WT: [n_cols,256] bf16 (pre-transposed weights)
// Double-buffered LDS, tiles staged with async-to-LDS DMA overlapped with WMMA.
template <int BN, bool ELU>
__global__ __launch_bounds__(256) void wmma_gemm(
    const __bf16* __restrict__ X, const __bf16* __restrict__ WT,
    const float* __restrict__ bias, float* __restrict__ out,
    int n_rows, int n_cols) {
  constexpr int BM = 64;
  constexpr int LDA = 40;               // padded row stride (80B, 16B aligned)
  constexpr int CT = BN / 32;           // col tiles per wave
  constexpr int NBC = BN / 64;          // B staging chunks per thread
  constexpr unsigned ABYTES = BM * LDA * 2;
  constexpr unsigned BBYTES = BN * LDA * 2;
  __shared__ __bf16 sA[2 * BM * LDA];
  __shared__ __bf16 sB[2 * BN * LDA];

  const int row0 = blockIdx.x * BM;
  const int col0 = blockIdx.y * BN;
  const int t = threadIdx.x;
  const int wave = t >> 5, lane = t & 31;
  const int rt = wave >> 1, cg = wave & 1;
  const int lm = lane & 15, lh = lane >> 4;

  v8f acc[CT];
  for (int c = 0; c < CT; ++c)
    for (int i = 0; i < 8; ++i) acc[c][i] = 0.f;

  // per-thread staging chunks: A = 256 x 16B covering 64x32 bf16
  int ar = t >> 2, ac = (t & 3) * 8;
  int agr = row0 + ar; if (agr >= n_rows) agr = n_rows - 1;   // clamp (never stored)
  const __bf16* gA = X + (size_t)agr * FDIM + ac;
  unsigned ldsA = lds_off32(&sA[ar * LDA + ac]);

  const __bf16* gB[NBC];
  unsigned ldsB[NBC];
#pragma unroll
  for (int i = 0; i < NBC; ++i) {
    int idx = t + i * 256;
    int br = idx >> 2, bc = (idx & 3) * 8;
    gB[i] = WT + (size_t)(col0 + br) * FDIM + bc;
    ldsB[i] = lds_off32(&sB[br * LDA + bc]);
  }

  // issue tile 0 into buffer 0
  async_ld_b128(ldsA, gA);
#pragma unroll
  for (int i = 0; i < NBC; ++i) async_ld_b128(ldsB[i], gB[i]);

  for (int kb = 0; kb < FDIM; kb += 32) {
    const int buf = (kb >> 5) & 1;
    wait_async0();          // this wave's copies for tile kb have landed
    __syncthreads();        // => every wave's copies for tile kb are in LDS,
                            //    and everyone finished reading buffer buf^1
    if (kb + 32 < FDIM) {   // stage tile kb+1 into the other buffer (overlaps compute)
      const int nb = buf ^ 1;
      async_ld_b128(ldsA + nb * ABYTES, gA + kb + 32);
#pragma unroll
      for (int i = 0; i < NBC; ++i)
        async_ld_b128(ldsB[i] + nb * BBYTES, gB[i] + kb + 32);
    }

    const __bf16* cA = sA + buf * (BM * LDA);
    const __bf16* cB = sB + buf * (BN * LDA);

    // A fragment: lane holds row rt*16+lm, K chunks lh*8 and 16+lh*8
    v8bf alo = *(const v8bf*)&cA[(rt * 16 + lm) * LDA + lh * 8];
    v8bf ahi = *(const v8bf*)&cA[(rt * 16 + lm) * LDA + 16 + lh * 8];
    v16bf a = __builtin_shufflevector(alo, ahi, 0, 1, 2, 3, 4, 5, 6, 7,
                                      8, 9, 10, 11, 12, 13, 14, 15);
#pragma unroll
    for (int c = 0; c < CT; ++c) {
      int bn = (cg * CT + c) * 16 + lm;   // output column within tile
      v8bf blo = *(const v8bf*)&cB[bn * LDA + lh * 8];
      v8bf bhi = *(const v8bf*)&cB[bn * LDA + 16 + lh * 8];
      v16bf b = __builtin_shufflevector(blo, bhi, 0, 1, 2, 3, 4, 5, 6, 7,
                                        8, 9, 10, 11, 12, 13, 14, 15);
      acc[c] = __builtin_amdgcn_wmma_f32_16x16x32_bf16(
          false, a, false, b, (short)0, acc[c], false, false);
    }
  }

#pragma unroll
  for (int c = 0; c < CT; ++c) {
    int col = col0 + (cg * CT + c) * 16 + lm;
    float bv = (col < n_cols) ? bias[col] : 0.f;
#pragma unroll
    for (int v = 0; v < 8; ++v) {
      int gr = row0 + rt * 16 + lh * 8 + v;
      if (gr < n_rows && col < n_cols) {
        float x = acc[c][v] + bv;
        if (ELU) x = eluf(x);
        out[(size_t)gr * n_cols + col] = x;
      }
    }
  }
}

// ------------------------------------------------- performer attention passes
// S layout: [n,512] = q(0:128)|k(128:256)|v(256:384)|gnn1(384:512)
__global__ __launch_bounds__(256) void k_headmax(const float* __restrict__ S,
                                                 const float* __restrict__ proj,
                                                 float* __restrict__ hm, int n_rows) {
  __shared__ float sp[NB * HD];
  for (int i = threadIdx.x; i < NB * HD; i += 256) sp[i] = proj[i];
  __syncthreads();
  int h = threadIdx.x & 7, slot = threadIdx.x >> 3;
  float mx = -3.4e38f;
  for (int n = blockIdx.x * 32 + slot; n < n_rows; n += gridDim.x * 32) {
    const float* kr = S + (size_t)n * 512 + 128 + h * HD;
    float kv[HD];
#pragma unroll
    for (int d = 0; d < HD; ++d) kv[d] = kr[d] * 0.5f;
    for (int m = 0; m < NB; ++m) {
      float dot = 0.f;
#pragma unroll
      for (int d = 0; d < HD; ++d) dot += kv[d] * sp[m * HD + d];
      mx = fmaxf(mx, dot);
    }
  }
  atomicMaxFloat(&hm[h], mx);
}

__global__ __launch_bounds__(256) void k_kp_accum(
    const float* __restrict__ S, const float* __restrict__ proj,
    const float* __restrict__ hm, float* __restrict__ kcum,
    float* __restrict__ ctx, int n_rows) {
  __shared__ float sp[NB * HD];
  __shared__ float skc[NHEAD * NB];
  __shared__ float sctx[NHEAD * NB * HD];
  for (int i = threadIdx.x; i < NB * HD; i += 256) sp[i] = proj[i];
  for (int i = threadIdx.x; i < NHEAD * NB; i += 256) skc[i] = 0.f;
  for (int i = threadIdx.x; i < NHEAD * NB * HD; i += 256) sctx[i] = 0.f;
  __syncthreads();
  int h = threadIdx.x & 7, slot = threadIdx.x >> 3;
  float M = hm[h];
  for (int n = blockIdx.x * 32 + slot; n < n_rows; n += gridDim.x * 32) {
    const float* kr = S + (size_t)n * 512 + 128 + h * HD;
    const float* vr = S + (size_t)n * 512 + 256 + h * HD;
    float kv[HD], vv[HD], diag = 0.f;
#pragma unroll
    for (int d = 0; d < HD; ++d) { float x = kr[d]; kv[d] = x * 0.5f; diag += x * x; }
    diag *= 0.125f;   // 0.5 * normalizer^2
#pragma unroll
    for (int d = 0; d < HD; ++d) vv[d] = vr[d];
    for (int m = 0; m < NB; ++m) {
      float dot = 0.f;
#pragma unroll
      for (int d = 0; d < HD; ++d) dot += kv[d] * sp[m * HD + d];
      float kp = RATIO * (__expf(dot - diag - M) + 1e-4f);
      atomicAdd(&skc[h * NB + m], kp);
#pragma unroll
      for (int d = 0; d < HD; ++d) atomicAdd(&sctx[(h * NB + m) * HD + d], kp * vv[d]);
    }
  }
  __syncthreads();
  for (int i = threadIdx.x; i < NHEAD * NB; i += 256) atomicAdd(&kcum[i], skc[i]);
  for (int i = threadIdx.x; i < NHEAD * NB * HD; i += 256) atomicAdd(&ctx[i], sctx[i]);
}

// per (node,head): qp, d_inv, attention output; writes H = bf16 elu(concat(gnn1,attn))
__global__ __launch_bounds__(256) void k_attn_out(
    const float* __restrict__ S, const float* __restrict__ proj,
    const float* __restrict__ kcum, const float* __restrict__ ctx,
    __bf16* __restrict__ H, int n_rows) {
  __shared__ float sp[NB * HD];
  __shared__ float skc[NHEAD * NB];
  __shared__ float sctx[NHEAD * NB * HD];
  for (int i = threadIdx.x; i < NB * HD; i += 256) sp[i] = proj[i];
  for (int i = threadIdx.x; i < NHEAD * NB; i += 256) skc[i] = kcum[i];
  for (int i = threadIdx.x; i < NHEAD * NB * HD; i += 256) sctx[i] = ctx[i];
  __syncthreads();
  int g = blockIdx.x * 256 + threadIdx.x;
  int n = g >> 3, h = g & 7;
  if (n >= n_rows) return;
  const float* qr = S + (size_t)n * 512 + h * HD;
  float qv[HD], diag = 0.f;
#pragma unroll
  for (int d = 0; d < HD; ++d) { float x = qr[d]; qv[d] = x * 0.5f; diag += x * x; }
  diag *= 0.125f;
  float dash[NB], mx = -3.4e38f;
  for (int m = 0; m < NB; ++m) {
    float dot = 0.f;
#pragma unroll
    for (int d = 0; d < HD; ++d) dot += qv[d] * sp[m * HD + d];
    dash[m] = dot; mx = fmaxf(mx, dot);
  }
  float den = 0.f;
  for (int m = 0; m < NB; ++m) {
    float qp = RATIO * (__expf(dash[m] - diag - mx) + 1e-4f);
    dash[m] = qp;
    den += qp * skc[h * NB + m];
  }
  float dinv = 1.f / den;
  const float* gr = S + (size_t)n * 512 + 384 + h * HD;
  for (int d = 0; d < HD; ++d) {
    float o = 0.f;
    for (int m = 0; m < NB; ++m) o += dash[m] * sctx[(h * NB + m) * HD + d];
    H[(size_t)n * 256 + 128 + h * HD + d] = f2bf(eluf(o * dinv));
    H[(size_t)n * 256 + h * HD + d] = f2bf(eluf(gr[d]));
  }
}

// ------------------------------------------------- APPNP
__global__ __launch_bounds__(256) void k_deg(const int* __restrict__ ei,
                                             float* __restrict__ deg, int E) {
  int e = blockIdx.x * 256 + threadIdx.x;
  if (e < E) atomicAdd(&deg[ei[(size_t)E + e]], 1.f);
}
__global__ __launch_bounds__(256) void k_dis(const float* __restrict__ deg,
                                             float* __restrict__ dis, int n) {
  int i = blockIdx.x * 256 + threadIdx.x;
  if (i < n) dis[i] = rsqrtf(deg[i] + 1.f);   // +1 self loop
}
__global__ __launch_bounds__(256) void k_hop_init(const float* __restrict__ logits,
                                                  const float* __restrict__ dis,
                                                  float* __restrict__ h,
                                                  float* __restrict__ agg, int n) {
  size_t g = (size_t)blockIdx.x * 256 + threadIdx.x;
  if (g >= (size_t)n * NCLS) return;
  int node = (int)(g >> 6);
  float d2 = dis[node] * dis[node];
  float x = logits[g];
  h[g] = x;
  agg[g] = d2 * x;   // self-loop contribution
}
__global__ __launch_bounds__(256) void k_scatter(const int* __restrict__ ei,
                                                 const float* __restrict__ dis,
                                                 const float* __restrict__ h,
                                                 float* __restrict__ agg, int E) {
  size_t g = (size_t)blockIdx.x * 256 + threadIdx.x;
  int e = (int)(g >> 2);
  if (e >= E) return;
  int q = ((int)g & 3) * 16;
  int r = ei[e], c = ei[(size_t)E + e];
  float nm = dis[r] * dis[c];
  const float4* hr = (const float4*)(h + (size_t)r * NCLS + q);
  float* ag = agg + (size_t)c * NCLS + q;
#pragma unroll
  for (int i = 0; i < 4; ++i) {
    float4 v = hr[i];
    atomicAdd(ag + i * 4 + 0, nm * v.x);
    atomicAdd(ag + i * 4 + 1, nm * v.y);
    atomicAdd(ag + i * 4 + 2, nm * v.z);
    atomicAdd(ag + i * 4 + 3, nm * v.w);
  }
}
__global__ __launch_bounds__(256) void k_hop_fin(const float* __restrict__ logits,
                                                 const float* __restrict__ dis,
                                                 float* __restrict__ h,
                                                 float* __restrict__ agg, int n) {
  size_t g = (size_t)blockIdx.x * 256 + threadIdx.x;
  if (g >= (size_t)n * NCLS) return;
  int node = (int)(g >> 6);
  float x = 0.9f * agg[g] + 0.1f * logits[g];
  h[g] = x;
  agg[g] = dis[node] * dis[node] * x;   // re-init self term for next hop
}
__global__ __launch_bounds__(256) void k_logsoftmax(const float* __restrict__ h,
                                                    float* __restrict__ out, int n) {
  int node = blockIdx.x * 256 + threadIdx.x;
  if (node >= n) return;
  const float* r = h + (size_t)node * NCLS;
  float mx = -3.4e38f;
  for (int c = 0; c < NCLS; ++c) mx = fmaxf(mx, r[c]);
  float s = 0.f;
  for (int c = 0; c < NCLS; ++c) s += __expf(r[c] - mx);
  float ls = logf(s) + mx;
  for (int c = 0; c < NCLS; ++c) {
    out[(size_t)node * NCLS + c] = r[c] - ls;
    out[(size_t)n * NCLS + (size_t)node * NCLS + c] = r[c];
  }
}

// ----------------------------------------------------------------- launcher
extern "C" void kernel_launch(void* const* d_in, const int* in_sizes, int n_in,
                              void* d_out, int out_size, void* d_ws, size_t ws_size,
                              hipStream_t stream) {
  const float* data   = (const float*)d_in[0];
  const int*   ei     = (const int*)d_in[1];
  const float* lin1_w = (const float*)d_in[2];
  const float* lin1_b = (const float*)d_in[3];
  const float* lin2_w = (const float*)d_in[4];
  const float* lin2_b = (const float*)d_in[5];
  const float* q_w = (const float*)d_in[6],  *q_b = (const float*)d_in[7];
  const float* k_w = (const float*)d_in[8],  *k_b = (const float*)d_in[9];
  const float* v_w = (const float*)d_in[10], *v_b = (const float*)d_in[11];
  const float* proj = (const float*)d_in[12];

  const int n = in_sizes[0] / FDIM;
  const int E = in_sizes[1] / 2;

  char* ws = (char*)d_ws;
  size_t off = 0;
  auto alloc = [&](size_t bytes) -> char* {
    char* p = ws + off;
    off = (off + bytes + 255) & ~(size_t)255;
    return p;
  };
  __bf16* XB  = (__bf16*)alloc((size_t)n * FDIM * 2);
  __bf16* WT1 = (__bf16*)alloc((size_t)512 * FDIM * 2);
  __bf16* WT2 = (__bf16*)alloc((size_t)NCLS * FDIM * 2);
  float*  B1  = (float*)alloc(512 * 4);
  float*  B2  = (float*)alloc(NCLS * 4);
  float*  HM  = (float*)alloc(NHEAD * 4);
  float*  KC  = (float*)alloc(NHEAD * NB * 4);
  float*  CTX = (float*)alloc(NHEAD * NB * HD * 4);
  __bf16* H   = (__bf16*)alloc((size_t)n * 256 * 2);
  float*  S   = (float*)alloc((size_t)n * 512 * 4);
  // overlay APPNP buffers onto S (S dead after k_attn_out)
  float* LOG = S;
  float* HB  = S + (size_t)n * 64;
  float* AG  = S + (size_t)n * 128;
  float* DEG = S + (size_t)n * 192;
  float* DIS = DEG + n;

  size_t tot = (size_t)n * FDIM;
  k_cvt_data<<<dim3((unsigned)((tot + 255) / 256)), dim3(256), 0, stream>>>(data, XB, tot);
  k_pack_w<<<dim3(512), dim3(256), 0, stream>>>(q_w, k_w, v_w, lin1_w,
                                                q_b, k_b, v_b, lin1_b, WT1, B1);
  k_pack_w2<<<dim3(NCLS), dim3(256), 0, stream>>>(lin2_w, lin2_b, WT2, B2);
  k_zero_small<<<dim3((NHEAD * NB * HD + 255) / 256), dim3(256), 0, stream>>>(HM, KC, CTX);

  dim3 g1((n + 63) / 64, 4);
  wmma_gemm<128, false><<<g1, dim3(256), 0, stream>>>(XB, WT1, B1, S, n, 512);

  k_headmax<<<dim3(256), dim3(256), 0, stream>>>(S, proj, HM, n);
  k_kp_accum<<<dim3(256), dim3(256), 0, stream>>>(S, proj, HM, KC, CTX, n);
  k_attn_out<<<dim3((n * 8 + 255) / 256), dim3(256), 0, stream>>>(S, proj, KC, CTX, H, n);

  dim3 g2((n + 63) / 64, 1);
  wmma_gemm<64, true><<<g2, dim3(256), 0, stream>>>(H, WT2, B2, LOG, n, NCLS);

  k_zero_f<<<dim3((n + 255) / 256), dim3(256), 0, stream>>>(DEG, n);
  k_deg<<<dim3((E + 255) / 256), dim3(256), 0, stream>>>(ei, DEG, E);
  k_dis<<<dim3((n + 255) / 256), dim3(256), 0, stream>>>(DEG, DIS, n);

  unsigned gNC = (unsigned)(((size_t)n * NCLS + 255) / 256);
  k_hop_init<<<dim3(gNC), dim3(256), 0, stream>>>(LOG, DIS, HB, AG, n);
  unsigned gE4 = (unsigned)(((size_t)E * 4 + 255) / 256);
  for (int hop = 0; hop < KHOPS; ++hop) {
    k_scatter<<<dim3(gE4), dim3(256), 0, stream>>>(ei, DIS, HB, AG, E);
    k_hop_fin<<<dim3(gNC), dim3(256), 0, stream>>>(LOG, DIS, HB, AG, n);
  }
  k_logsoftmax<<<dim3((n + 255) / 256), dim3(256), 0, stream>>>(HB, (float*)d_out, n);
  (void)n_in; (void)out_size; (void)ws_size;
}
